// RNN_CrossAttention_4458176053695
// MI455X (gfx1250) — matrix operand order repaired and checked by reading, not compiled, not run on hardware
//
#include <hip/hip_runtime.h>
#include <hip/hip_bf16.h>

// Problem dims (fixed by the reference)
#define B_ 32
#define T_ 128
#define U_ 768
#define D_ 768

typedef __attribute__((ext_vector_type(16))) __bf16 v16bf;
typedef __attribute__((ext_vector_type(8)))  float  v8f;

typedef uint32_t u32x4 __attribute__((ext_vector_type(4)));
typedef int32_t  i32x8 __attribute__((ext_vector_type(8)));
typedef int32_t  i32x4 __attribute__((ext_vector_type(4)));

static __device__ __forceinline__ __bf16 f2bf(float f) { return (__bf16)f; }

// LDS byte offset of a pointer into a __shared__ object (generic -> AS(3))
static __device__ __forceinline__ uint32_t lds_off(const void* p) {
  return (uint32_t)(uintptr_t)(const __attribute__((address_space(3))) char*)p;
}

// ---------------------------------------------------------------------------
// Tensor Data Mover: DMA a [dim1 x dim0] fp32 tile (row stride = stride0
// elements) from global into LDS at lds_byte_off. D# layout per ISA ch.8:
//   group0: count=1 | lds_addr | global_addr[56:0] | type=2
//   group1: data_size=4B, tensor_dim0/1, tile_dim0/1, tensor_dim0_stride
//   groups 2/3 (+ extra group on this toolchain): zero (<=2D tensor)
// workgroup_mask = 0 (not in a cluster). Tracked by TENSORcnt.
// ---------------------------------------------------------------------------
static __device__ __forceinline__ void tdm_load_2d(uint32_t lds_byte_off,
                                                   const void* gaddr,
                                                   uint32_t dim0, uint32_t dim1,
                                                   uint32_t stride0) {
  uint64_t ga = (uint64_t)(uintptr_t)gaddr;
  u32x4 g0;
  g0[0] = 1u;                                               // count=1 (valid)
  g0[1] = lds_byte_off;                                     // lds_addr
  g0[2] = (uint32_t)ga;                                     // global_addr lo
  g0[3] = (uint32_t)((ga >> 32) & 0x01FFFFFFu) | (2u << 30);// addr hi | type=2
  i32x8 g1;
  g1[0] = (int)(2u << 16);                                  // data_size=4B
  g1[1] = (int)((dim0 & 0xFFFFu) << 16);                    // tensor_dim0 lo16
  g1[2] = (int)(((dim0 >> 16) & 0xFFFFu) | ((dim1 & 0xFFFFu) << 16));
  g1[3] = (int)(((dim1 >> 16) & 0xFFFFu) | ((dim0 & 0xFFFFu) << 16)); // tile_dim0
  g1[4] = (int)(dim1 & 0xFFFFu);                            // tile_dim1, tile_dim2=0
  g1[5] = (int)stride0;                                     // dim0 stride lo32
  g1[6] = 0;                                                // stride hi, dim1_stride lo
  g1[7] = 0;
  i32x4 z4 = {0, 0, 0, 0};
  i32x8 z8 = {0, 0, 0, 0, 0, 0, 0, 0};
  __builtin_amdgcn_tensor_load_to_lds(g0, g1, z4, z4, z8, 0);
}

// ---------------------------------------------------------------------------
// h := h0   (re-done every call -> deterministic)
// ---------------------------------------------------------------------------
__global__ void init_h(const float* __restrict__ h0, float* __restrict__ h) {
  int i = blockIdx.x * blockDim.x + threadIdx.x;
  if (i < B_ * U_) h[i] = h0[i];
}

// ---------------------------------------------------------------------------
// pre[b,u] = bias[u] + sum_d x[b,t,d]*kernel[u,d] + sum_v h[b,v]*rk[u,v]
// One wave per 16(b) x 16(u) tile, V_WMMA_F32_16X16X32_BF16.
// A (16x32 bf16): lane L: M=L&15, half=L>>4; K=half*16+e
// B (32x16 bf16): lane L: N=L&15, K=(L>>4)*16+e  (kernel row-major => contiguous)
// C/D (16x16 f32): VGPR r, lane L -> M=(L>>4)*8+r, N=L&15
// ---------------------------------------------------------------------------
__global__ void dense_gemm_step(const float* __restrict__ x,
                                const float* __restrict__ h,
                                const float* __restrict__ kern,
                                const float* __restrict__ rk,
                                const float* __restrict__ bias,
                                float* __restrict__ pre,
                                int t) {
  const int lane = threadIdx.x;        // blockDim.x == 32: EXEC all-1s for WMMA
  const int m    = lane & 15;
  const int half = lane >> 4;
  const int u0   = blockIdx.x * 16;
  const int b0   = blockIdx.y * 16;

  const float* xrow  = x    + ((size_t)(b0 + m) * T_ + t) * D_;
  const float* hrow  = h    + (size_t)(b0 + m) * U_;
  const float* krow  = kern + (size_t)(u0 + m) * D_;
  const float* rkrow = rk   + (size_t)(u0 + m) * U_;

  v8f c = {};

  for (int kb = 0; kb < D_; kb += 32) {          // x_t @ kernel.T
    v16bf a, bm;
    const float* ap0 = xrow + kb + half * 16;
    const float* ap1 = xrow + kb + half * 16 + 8;
#pragma unroll
    for (int e = 0; e < 8; ++e) { a[e] = f2bf(ap0[e]); a[8 + e] = f2bf(ap1[e]); }
    const float* bp = krow + kb + half * 16;
#pragma unroll
    for (int e = 0; e < 16; ++e) bm[e] = f2bf(bp[e]);
    c = __builtin_amdgcn_wmma_f32_16x16x32_bf16(false, a, false, bm,
                                                (short)0, c, false, false);
  }
  for (int kb = 0; kb < U_; kb += 32) {          // h @ rk.T
    v16bf a, bm;
    const float* ap0 = hrow + kb + half * 16;
    const float* ap1 = hrow + kb + half * 16 + 8;
#pragma unroll
    for (int e = 0; e < 8; ++e) { a[e] = f2bf(ap0[e]); a[8 + e] = f2bf(ap1[e]); }
    const float* bp = rkrow + kb + half * 16;
#pragma unroll
    for (int e = 0; e < 16; ++e) bm[e] = f2bf(bp[e]);
    c = __builtin_amdgcn_wmma_f32_16x16x32_bf16(false, a, false, bm,
                                                (short)0, c, false, false);
  }

  const float bs   = bias[u0 + m];
  const int   rowb = b0 + half * 8;
#pragma unroll
  for (int r = 0; r < 8; ++r)
    pre[(size_t)(rowb + r) * U_ + (u0 + m)] = c[r] + bs;
}

// ---------------------------------------------------------------------------
// Attention step. Block = 8 waves, one (b,u) per wave. TDM stages the shared
// x[b,t,:], h[b,:] rows and the 8-row ak/rak tiles into LDS (54 KB), then the
// fused exp/normalizer/numerator loops run float4-vectorized out of LDS + L2.
// ---------------------------------------------------------------------------
__global__ void attn_step(const float* __restrict__ x,
                          const float* __restrict__ h,
                          const float* __restrict__ kern,
                          const float* __restrict__ ak,
                          const float* __restrict__ rk,
                          const float* __restrict__ rak,
                          const float* __restrict__ pre,
                          const float* __restrict__ gpa,
                          const float* __restrict__ gpra,
                          float* __restrict__ out,
                          float* __restrict__ hnext,
                          int t) {
  __shared__ float xs[D_];          //  3 KB  x[b,t,:]
  __shared__ float hs[U_];          //  3 KB  h[b,:]
  __shared__ float aks[8 * D_];     // 24 KB  ak[u0..u0+7,:]
  __shared__ float raks[8 * U_];    // 24 KB  rak[u0..u0+7,:]

  const int wave = threadIdx.x >> 5;           // 0..7
  const int lane = threadIdx.x & 31;
  const int u0   = blockIdx.x * 8;
  const int u    = u0 + wave;
  const int b    = blockIdx.y;

  const float* xrow = x + ((size_t)b * T_ + t) * D_;
  const float* hrow = h + (size_t)b * U_;

  if (threadIdx.x == 0) {
    tdm_load_2d(lds_off(xs),   xrow,                    D_, 1, D_);
    tdm_load_2d(lds_off(hs),   hrow,                    U_, 1, U_);
    tdm_load_2d(lds_off(aks),  ak  + (size_t)u0 * D_,   D_, 8, D_);
    tdm_load_2d(lds_off(raks), rak + (size_t)u0 * U_,   U_, 8, U_);
    __builtin_amdgcn_s_wait_tensorcnt(0);
  }
  __syncthreads();

  const float4* xs4   = reinterpret_cast<const float4*>(xs);
  const float4* hs4   = reinterpret_cast<const float4*>(hs);
  const float4* aks4  = reinterpret_cast<const float4*>(aks)  + wave * (D_ / 4);
  const float4* raks4 = reinterpret_cast<const float4*>(raks) + wave * (U_ / 4);
  const float4* krow4  = reinterpret_cast<const float4*>(kern + (size_t)u * D_);
  const float4* rkrow4 = reinterpret_cast<const float4*>(rk   + (size_t)u * U_);

  __builtin_prefetch(krow4 + lane * 2, 0, 0);   // warm L2 path
  __builtin_prefetch(rkrow4 + lane * 2, 0, 0);

  const float hbu = hs[u];
  float s1 = 0.f, n1 = 0.f, s2 = 0.f, n2 = 0.f;

  for (int i = lane; i < D_ / 4; i += 32) {
    float4 xv = xs4[i];
    float4 av = aks4[i];
    float4 kv = krow4[i];
    float e0 = __expf(hbu * xv.x * av.x);
    float e1 = __expf(hbu * xv.y * av.y);
    float e2 = __expf(hbu * xv.z * av.z);
    float e3 = __expf(hbu * xv.w * av.w);
    s1 += (e0 + e1) + (e2 + e3);
    n1 = __fmaf_rn(e0 * kv.x, xv.x, n1);
    n1 = __fmaf_rn(e1 * kv.y, xv.y, n1);
    n1 = __fmaf_rn(e2 * kv.z, xv.z, n1);
    n1 = __fmaf_rn(e3 * kv.w, xv.w, n1);
  }
  for (int i = lane; i < U_ / 4; i += 32) {
    float4 hv = hs4[i];
    float4 av = raks4[i];
    float4 kv = rkrow4[i];
    float e0 = __expf(hbu * hv.x * av.x);
    float e1 = __expf(hbu * hv.y * av.y);
    float e2 = __expf(hbu * hv.z * av.z);
    float e3 = __expf(hbu * hv.w * av.w);
    s2 += (e0 + e1) + (e2 + e3);
    n2 = __fmaf_rn(e0 * kv.x, hv.x, n2);
    n2 = __fmaf_rn(e1 * kv.y, hv.y, n2);
    n2 = __fmaf_rn(e2 * kv.z, hv.z, n2);
    n2 = __fmaf_rn(e3 * kv.w, hv.w, n2);
  }

#pragma unroll
  for (int off = 16; off > 0; off >>= 1) {
    s1 += __shfl_xor(s1, off, 32);
    n1 += __shfl_xor(n1, off, 32);
    s2 += __shfl_xor(s2, off, 32);
    n2 += __shfl_xor(n2, off, 32);
  }

  if (lane == 0) {
    float ga  = 1.f / (1.f + __expf(-gpa[0]));
    float gra = 1.f / (1.f + __expf(-gpra[0]));
    float o = pre[(size_t)b * U_ + u] + ga * n1 / s1 + gra * n2 / s2;
    o = tanhf(o);
    out[((size_t)b * T_ + t) * U_ + u] = o;
    hnext[(size_t)b * U_ + u] = o;
  }
}

// ---------------------------------------------------------------------------
// Orchestration: init h, then 128 steps of (WMMA GEMM, TDM+exp attention),
// ping-ponging h in workspace. All launches on `stream` (graph-capture safe).
// ---------------------------------------------------------------------------
extern "C" void kernel_launch(void* const* d_in, const int* in_sizes, int n_in,
                              void* d_out, int out_size, void* d_ws, size_t ws_size,
                              hipStream_t stream) {
  (void)in_sizes; (void)n_in; (void)out_size; (void)ws_size;

  const float* x    = (const float*)d_in[0];  // [B,T,D]
  const float* h0   = (const float*)d_in[1];  // [B,U]
  const float* kern = (const float*)d_in[2];  // [U,D]
  const float* rk   = (const float*)d_in[3];  // [U,U]
  const float* ak   = (const float*)d_in[4];  // [U,D]
  const float* rak  = (const float*)d_in[5];  // [U,U]
  const float* bias = (const float*)d_in[6];  // [U]
  const float* gpa  = (const float*)d_in[7];  // scalar
  const float* gpra = (const float*)d_in[8];  // scalar
  float* out = (float*)d_out;                 // [B,T,U]

  float* hA  = (float*)d_ws;                  // [B,U]
  float* hB  = hA + (size_t)B_ * U_;          // [B,U]
  float* pre = hB + (size_t)B_ * U_;          // [B,U]

  init_h<<<(B_ * U_ + 255) / 256, 256, 0, stream>>>(h0, hA);

  for (int t = 0; t < T_; ++t) {
    const float* hc = (t & 1) ? hB : hA;
    float*       hn = (t & 1) ? hA : hB;
    dense_gemm_step<<<dim3(U_ / 16, B_ / 16), 32, 0, stream>>>(
        x, hc, kern, rk, bias, pre, t);
    attn_step<<<dim3(U_ / 8, B_), 256, 0, stream>>>(
        x, hc, kern, ak, rk, rak, pre, gpa, gpra, out, hn, t);
  }
}
